// MultiHeadAttention_28570122453217
// MI455X (gfx1250) — matrix operand order repaired and checked by reading, not compiled, not run on hardware
//
#include <hip/hip_runtime.h>

typedef __bf16 bf16;
typedef __attribute__((ext_vector_type(8)))  __bf16 v8bf;
typedef __attribute__((ext_vector_type(16))) __bf16 v16bf;
typedef __attribute__((ext_vector_type(8)))  float  v8f;

#define MODEL_DIM 1024
#define SEQ_L     2048
#define BATCH     16
#define NROWS     (BATCH * SEQ_L)          // 32768
#define ATTN_SCALE 0.08838834764831845f    // (1024/8)^-0.5

// ---------------------------------------------------------------------------
// CDNA5 async global->LDS staging (ASYNCcnt-tracked), with safe fallback.
// Builtin signature (from clang diagnostic): param0 = int4 ptr in global AS,
// param1 = int4 ptr in LDS AS, then imm offset, imm cpol.
// ---------------------------------------------------------------------------
#if __has_builtin(__builtin_amdgcn_global_load_async_to_lds_b128) && \
    __has_builtin(__builtin_amdgcn_s_wait_asynccnt)
#define USE_ASYNC_LDS 1
typedef int v4i_vec __attribute__((vector_size(16)));
typedef __attribute__((address_space(1))) v4i_vec as1_v4i;
typedef __attribute__((address_space(3))) v4i_vec as3_v4i;
#else
#define USE_ASYNC_LDS 0
#endif

static __device__ __forceinline__ v8bf ld8(const bf16* p) { return *(const v8bf*)p; }

static __device__ __forceinline__ void cp16_g2l(bf16* dst_lds, const bf16* src) {
#if USE_ASYNC_LDS
  __builtin_amdgcn_global_load_async_to_lds_b128((as1_v4i*)src, (as3_v4i*)dst_lds, 0, 0);
#else
  *(v8bf*)dst_lds = *(const v8bf*)src;
#endif
}

static __device__ __forceinline__ void async_wait_all() {
#if USE_ASYNC_LDS
  __builtin_amdgcn_s_wait_asynccnt(0);
#endif
}

static __device__ __forceinline__ v16bf cat8(v8bf lo, v8bf hi) {
  return __builtin_shufflevector(lo, hi, 0,1,2,3,4,5,6,7,8,9,10,11,12,13,14,15);
}

static __device__ __forceinline__ v8f wmma_bf16(v16bf a, v16bf b, v8f c) {
  // (neg_a, A, neg_b, B, c_mod, C, reuse_a, reuse_b)
  return __builtin_amdgcn_wmma_f32_16x16x32_bf16(false, a, false, b, (short)0, c, false, false);
}

// ---------------------------------------------------------------------------
// Weight transpose + fp32->bf16 convert: Wt[n][k] = (bf16)W[k][n]
// ---------------------------------------------------------------------------
__global__ __launch_bounds__(256) void wconv_kernel(const float* __restrict__ W,
                                                    bf16* __restrict__ Wt) {
  __shared__ float tile[32][33];
  const int tx = threadIdx.x & 31;
  const int ty = threadIdx.x >> 5;       // 0..7
  const int c0 = blockIdx.x * 32;        // n
  const int r0 = blockIdx.y * 32;        // k
  #pragma unroll
  for (int i = ty; i < 32; i += 8)
    tile[i][tx] = W[(size_t)(r0 + i) * MODEL_DIM + c0 + tx];
  __syncthreads();
  #pragma unroll
  for (int i = ty; i < 32; i += 8)
    Wt[(size_t)(c0 + i) * MODEL_DIM + r0 + tx] = (bf16)tile[tx][i];
}

// ---------------------------------------------------------------------------
// Projection GEMM: out = A_f32[M,1024] * W + bias (Wt pre-transposed bf16)
// STORE_T==0: out[row][col] bf16   STORE_T==1: out transposed Vt[b][col][l]
// Block 128x128, 8 waves (4Mx2N), wave 32x64. Double-buffered K=32 tiles.
// ---------------------------------------------------------------------------
template <int STORE_T>
__global__ __launch_bounds__(256) void proj_gemm_kernel(const float* __restrict__ A,
                                                        const bf16* __restrict__ Wt,
                                                        const float* __restrict__ bias,
                                                        bf16* __restrict__ out) {
  __shared__ bf16 Alds[2][128 * 40];
  __shared__ bf16 Blds[2][128 * 40];
  const int tid  = threadIdx.x;
  const int wid  = tid >> 5;
  const int lane = tid & 31;
  const int mblk = blockIdx.y * 128;
  const int nblk = blockIdx.x * 128;
  const int wm   = (wid >> 1) * 32;
  const int wn   = (wid & 1) * 64;
  const int sr   = tid >> 1;            // staging row 0..127
  const int sh   = (tid & 1) * 16;      // staging half of 32-wide K slice
  const int arow = wm + (lane & 15);
  const int brow = wn + (lane & 15);
  const int kA   = (lane < 16) ? 0 : 8;
  const int kB   = (lane < 16) ? 0 : 8;

  const float* srcA0 = A  + (size_t)(mblk + sr) * MODEL_DIM + sh;
  const bf16*  srcB0 = Wt + (size_t)(nblk + sr) * MODEL_DIM + sh;

  auto stageA = [&](int buf, int k0) {  // fp32 -> bf16 convert (VALU path)
    const float* src = srcA0 + k0;
    alignas(16) float t[16];
    ((float4*)t)[0] = ((const float4*)src)[0];
    ((float4*)t)[1] = ((const float4*)src)[1];
    ((float4*)t)[2] = ((const float4*)src)[2];
    ((float4*)t)[3] = ((const float4*)src)[3];
    bf16* dst = &Alds[buf][sr * 40 + sh];
    #pragma unroll
    for (int i = 0; i < 16; ++i) dst[i] = (bf16)t[i];
  };
  auto stageB = [&](int buf, int k0) {  // bf16, async DMA path
    const bf16* src = srcB0 + k0;
    cp16_g2l(&Blds[buf][sr * 40 + sh],     src);
    cp16_g2l(&Blds[buf][sr * 40 + sh + 8], src + 8);
  };

  v8f acc[2][4] = {};
  stageA(0, 0);
  stageB(0, 0);
  int cur = 0;

  for (int k0 = 0; k0 < MODEL_DIM; k0 += 32) {
    async_wait_all();
    __syncthreads();
    if (k0 + 32 < MODEL_DIM) { stageA(cur ^ 1, k0 + 32); stageB(cur ^ 1, k0 + 32); }
    if (k0 + 64 < MODEL_DIM) {
      __builtin_prefetch(srcA0 + k0 + 64, 0, 0);
      __builtin_prefetch(srcB0 + k0 + 64, 0, 0);
    }

    v16bf a[2], b[4];
    #pragma unroll
    for (int i = 0; i < 2; ++i) {
      const bf16* p = &Alds[cur][(arow + i * 16) * 40 + kA];
      a[i] = cat8(ld8(p), ld8(p + 16));
    }
    #pragma unroll
    for (int j = 0; j < 4; ++j) {
      const bf16* p = &Blds[cur][(brow + j * 16) * 40 + kB];
      b[j] = cat8(ld8(p), ld8(p + 16));
    }
    #pragma unroll
    for (int i = 0; i < 2; ++i)
      #pragma unroll
      for (int j = 0; j < 4; ++j)
        acc[i][j] = wmma_bf16(a[i], b[j], acc[i][j]);
    cur ^= 1;
  }

  #pragma unroll
  for (int i = 0; i < 2; ++i)
    #pragma unroll
    for (int j = 0; j < 4; ++j) {
      const int col = nblk + wn + j * 16 + (lane & 15);
      const float bv = bias[col];
      if (STORE_T == 0) {
        #pragma unroll
        for (int r = 0; r < 8; ++r) {
          const int row = mblk + wm + i * 16 + r + ((lane >> 4) << 3);
          out[(size_t)row * MODEL_DIM + col] = (bf16)(acc[i][j][r] + bv);
        }
      } else {  // Vt[b][col][l]: 8 consecutive l per lane -> one b128 store
        const int row0 = mblk + wm + i * 16 + ((lane >> 4) << 3);
        const int bb = row0 >> 11;
        const int ll = row0 & (SEQ_L - 1);
        alignas(16) bf16 tmp[8];
        #pragma unroll
        for (int r = 0; r < 8; ++r) tmp[r] = (bf16)(acc[i][j][r] + bv);
        *(v8bf*)&out[((size_t)bb * MODEL_DIM + col) * SEQ_L + ll] = *(v8bf*)tmp;
      }
    }
}

// ---------------------------------------------------------------------------
// Flash-style attention: per (batch, 64 q-rows, 256 d-cols) block, 4 waves.
// Streams 32 keys/chunk with double-buffered K/V tiles (async staged).
// LDS: q 129KB + 2x64.5KB (K) + 2x20KB (Vt) + 5KB (P)  ~= 303KB of the 320KB WGP.
// ---------------------------------------------------------------------------
__global__ __launch_bounds__(128) void attn_kernel(const bf16* __restrict__ q,
                                                   const bf16* __restrict__ k,
                                                   const bf16* __restrict__ Vt,
                                                   bf16* __restrict__ ctx) {
  extern __shared__ char smem_raw[];
  bf16* q_lds   = (bf16*)smem_raw;           // 64 x 1032
  bf16* k_ldsA  = q_lds  + 64 * 1032;        // 32 x 1032
  bf16* k_ldsB  = k_ldsA + 32 * 1032;        // 32 x 1032
  bf16* vt_ldsA = k_ldsB + 32 * 1032;        // 256 x 40
  bf16* vt_ldsB = vt_ldsA + 256 * 40;        // 256 x 40
  bf16* p_lds   = vt_ldsB + 256 * 40;        // 64 x 40

  const int tid   = threadIdx.x;
  const int wid   = tid >> 5;
  const int lane  = tid & 31;
  const int qrow0 = blockIdx.x * 64;
  const int col0  = blockIdx.y * 256;
  const int bz    = blockIdx.z;
  const int mbase = (wid >> 1) * 32;   // this wave's q rows within block
  const int nwb   = (wid & 1) * 128;   // this wave's ctx columns within chunk

  // Park the whole 64x1024 q block in LDS once (async DMA).
  {
    const bf16* src = q + ((size_t)(bz * SEQ_L + qrow0)) * MODEL_DIM + tid * 8;
    #pragma unroll 4
    for (int v = 0; v < 64; ++v)
      cp16_g2l(&q_lds[v * 1032 + tid * 8], src + (size_t)v * MODEL_DIM);
  }

  auto stageK = [&](bf16* kl, int kb) {
    const bf16* src = k + ((size_t)(bz * SEQ_L + kb)) * MODEL_DIM + tid * 8;
    #pragma unroll 4
    for (int v = 0; v < 32; ++v)
      cp16_g2l(&kl[v * 1032 + tid * 8], src + (size_t)v * MODEL_DIM);
  };
  auto stageV = [&](bf16* vl, int kb) {
    #pragma unroll
    for (int c = 0; c < 2; ++c) {
      const int col = tid * 2 + c;
      const bf16* src = Vt + ((size_t)(bz * MODEL_DIM + col0 + col)) * SEQ_L + kb;
      bf16* d = &vl[col * 40];
      cp16_g2l(d,      src);
      cp16_g2l(d + 8,  src + 8);
      cp16_g2l(d + 16, src + 16);
      cp16_g2l(d + 24, src + 24);
    }
  };

  v8f cacc[2][8] = {};
  float m_st[16], l_st[16];
  #pragma unroll
  for (int i = 0; i < 16; ++i) { m_st[i] = -1e30f; l_st[i] = 0.0f; }

  const int kA    = (lane < 16) ? 0 : 8;
  const int kB    = (lane < 16) ? 0 : 8;
  const int arow0 = mbase + (lane & 15);

  stageK(k_ldsA, 0);
  stageV(vt_ldsA, 0);
  int cur = 0;

  for (int kb = 0; kb < SEQ_L; kb += 32) {
    async_wait_all();
    __syncthreads();
    bf16* kl = cur ? k_ldsB : k_ldsA;
    bf16* vl = cur ? vt_ldsB : vt_ldsA;
    if (kb + 32 < SEQ_L) {
      stageK(cur ? k_ldsA : k_ldsB, kb + 32);
      stageV(cur ? vt_ldsA : vt_ldsB, kb + 32);
    }
    if (kb + 64 < SEQ_L)
      __builtin_prefetch(k + ((size_t)(bz * SEQ_L + kb + 64)) * MODEL_DIM + tid * 8, 0, 0);

    // S tiles: this wave's 32 rows x 32 keys, K = 1024
    v8f sv[2][2] = {};
    for (int kd = 0; kd < MODEL_DIM; kd += 32) {
      const bf16* pa0 = &q_lds[arow0 * 1032 + kd + kA];
      const bf16* pa1 = &q_lds[(arow0 + 16) * 1032 + kd + kA];
      v16bf a0 = cat8(ld8(pa0), ld8(pa0 + 16));
      v16bf a1 = cat8(ld8(pa1), ld8(pa1 + 16));
      const bf16* pb0 = &kl[(lane & 15) * 1032 + kd + kB];
      const bf16* pb1 = &kl[((lane & 15) + 16) * 1032 + kd + kB];
      v16bf b0 = cat8(ld8(pb0), ld8(pb0 + 16));
      v16bf b1 = cat8(ld8(pb1), ld8(pb1 + 16));
      sv[0][0] = wmma_bf16(a0, b0, sv[0][0]);
      sv[0][1] = wmma_bf16(a0, b1, sv[0][1]);
      sv[1][0] = wmma_bf16(a1, b0, sv[1][0]);
      sv[1][1] = wmma_bf16(a1, b1, sv[1][1]);
    }

    // Online softmax over this 32-key chunk.
    #pragma unroll
    for (int i = 0; i < 2; ++i) {
      #pragma unroll
      for (int r = 0; r < 8; ++r) {
        const int slot = i * 8 + r;
        float s0 = sv[i][0][r] * ATTN_SCALE;
        float s1 = sv[i][1][r] * ATTN_SCALE;
        float rmax = fmaxf(s0, s1);
        #pragma unroll
        for (int off = 8; off; off >>= 1)
          rmax = fmaxf(rmax, __shfl_xor(rmax, off, 32));
        const float mnew = fmaxf(m_st[slot], rmax);
        const float corr = __expf(m_st[slot] - mnew);
        const float p0 = __expf(s0 - mnew);
        const float p1 = __expf(s1 - mnew);
        float rsum = p0 + p1;
        #pragma unroll
        for (int off = 8; off; off >>= 1)
          rsum += __shfl_xor(rsum, off, 32);
        l_st[slot] = l_st[slot] * corr + rsum;
        m_st[slot] = mnew;
        #pragma unroll
        for (int j = 0; j < 8; ++j) cacc[i][j][r] *= corr;
        if ((wid & 1) == 0) {  // one wave per row-half writes P (others duplicate)
          const int prow = mbase + i * 16 + r + ((lane >> 4) << 3);
          p_lds[prow * 40 + (lane & 15)]      = (bf16)p0;
          p_lds[prow * 40 + 16 + (lane & 15)] = (bf16)p1;
        }
      }
    }
    __syncthreads();

    // ctx += P(32x32) . V(32x128 of this wave's columns)
    {
      const bf16* pp0 = &p_lds[arow0 * 40 + kA];
      const bf16* pp1 = &p_lds[(arow0 + 16) * 40 + kA];
      v16bf pa0 = cat8(ld8(pp0), ld8(pp0 + 16));
      v16bf pa1 = cat8(ld8(pp1), ld8(pp1 + 16));
      #pragma unroll
      for (int j = 0; j < 8; ++j) {
        const bf16* pv = &vl[(nwb + j * 16 + (lane & 15)) * 40 + kB];
        v16bf vb = cat8(ld8(pv), ld8(pv + 16));
        cacc[0][j] = wmma_bf16(pa0, vb, cacc[0][j]);
        cacc[1][j] = wmma_bf16(pa1, vb, cacc[1][j]);
      }
    }
    cur ^= 1;
  }

  // Normalize and store ctx (bf16).
  float inv[16];
  #pragma unroll
  for (int i = 0; i < 16; ++i) inv[i] = 1.0f / l_st[i];
  #pragma unroll
  for (int i = 0; i < 2; ++i)
    #pragma unroll
    for (int j = 0; j < 8; ++j) {
      const int col = col0 + nwb + j * 16 + (lane & 15);
      #pragma unroll
      for (int r = 0; r < 8; ++r) {
        const int rowl = qrow0 + mbase + i * 16 + r + ((lane >> 4) << 3);
        ctx[((size_t)(bz * SEQ_L + rowl)) * MODEL_DIM + col] =
            (bf16)(cacc[i][j][r] * inv[i * 8 + r]);
      }
    }
}

// ---------------------------------------------------------------------------
// Output GEMM: out_f32 = ctx_bf16 * Wo + bo + residual(query)
// ---------------------------------------------------------------------------
__global__ __launch_bounds__(256) void out_gemm_kernel(const bf16* __restrict__ A,
                                                       const bf16* __restrict__ Wt,
                                                       const float* __restrict__ bias,
                                                       const float* __restrict__ resid,
                                                       float* __restrict__ out) {
  __shared__ bf16 Alds[2][128 * 40];
  __shared__ bf16 Blds[2][128 * 40];
  const int tid  = threadIdx.x;
  const int wid  = tid >> 5;
  const int lane = tid & 31;
  const int mblk = blockIdx.y * 128;
  const int nblk = blockIdx.x * 128;
  const int wm   = (wid >> 1) * 32;
  const int wn   = (wid & 1) * 64;
  const int sr   = tid >> 1;
  const int sh   = (tid & 1) * 16;
  const int arow = wm + (lane & 15);
  const int brow = wn + (lane & 15);
  const int kA   = (lane < 16) ? 0 : 8;
  const int kB   = (lane < 16) ? 0 : 8;

  const bf16* srcA0 = A  + (size_t)(mblk + sr) * MODEL_DIM + sh;
  const bf16* srcB0 = Wt + (size_t)(nblk + sr) * MODEL_DIM + sh;

  auto stage = [&](int buf, int k0) {
    cp16_g2l(&Alds[buf][sr * 40 + sh],     srcA0 + k0);
    cp16_g2l(&Alds[buf][sr * 40 + sh + 8], srcA0 + k0 + 8);
    cp16_g2l(&Blds[buf][sr * 40 + sh],     srcB0 + k0);
    cp16_g2l(&Blds[buf][sr * 40 + sh + 8], srcB0 + k0 + 8);
  };

  v8f acc[2][4] = {};
  stage(0, 0);
  int cur = 0;

  for (int k0 = 0; k0 < MODEL_DIM; k0 += 32) {
    async_wait_all();
    __syncthreads();
    if (k0 + 32 < MODEL_DIM) stage(cur ^ 1, k0 + 32);
    if (k0 + 64 < MODEL_DIM) {
      __builtin_prefetch(srcA0 + k0 + 64, 0, 0);
      __builtin_prefetch(srcB0 + k0 + 64, 0, 0);
    }

    v16bf a[2], b[4];
    #pragma unroll
    for (int i = 0; i < 2; ++i) {
      const bf16* p = &Alds[cur][(arow + i * 16) * 40 + kA];
      a[i] = cat8(ld8(p), ld8(p + 16));
    }
    #pragma unroll
    for (int j = 0; j < 4; ++j) {
      const bf16* p = &Blds[cur][(brow + j * 16) * 40 + kB];
      b[j] = cat8(ld8(p), ld8(p + 16));
    }
    #pragma unroll
    for (int i = 0; i < 2; ++i)
      #pragma unroll
      for (int j = 0; j < 4; ++j)
        acc[i][j] = wmma_bf16(a[i], b[j], acc[i][j]);
    cur ^= 1;
  }

  #pragma unroll
  for (int i = 0; i < 2; ++i)
    #pragma unroll
    for (int j = 0; j < 4; ++j) {
      const int col = nblk + wn + j * 16 + (lane & 15);
      const float bv = bias[col];
      #pragma unroll
      for (int r = 0; r < 8; ++r) {
        const int row = mblk + wm + i * 16 + r + ((lane >> 4) << 3);
        const size_t idx = (size_t)row * MODEL_DIM + col;
        out[idx] = acc[i][j][r] + bv + resid[idx];
      }
    }
}

// ---------------------------------------------------------------------------
extern "C" void kernel_launch(void* const* d_in, const int* in_sizes, int n_in,
                              void* d_out, int out_size, void* d_ws, size_t ws_size,
                              hipStream_t stream) {
  (void)in_sizes; (void)n_in; (void)out_size; (void)ws_size;
  const float* query = (const float*)d_in[0];
  const float* key_  = (const float*)d_in[1];
  const float* value = (const float*)d_in[2];
  const float* Wq = (const float*)d_in[3];
  const float* bq = (const float*)d_in[4];
  const float* Wk = (const float*)d_in[5];
  const float* bk = (const float*)d_in[6];
  const float* Wv = (const float*)d_in[7];
  const float* bv = (const float*)d_in[8];
  const float* Wo = (const float*)d_in[9];
  const float* bo = (const float*)d_in[10];
  float* out = (float*)d_out;

  char* ws = (char*)d_ws;
  size_t off = 0;
  const size_t WBYTES = (size_t)MODEL_DIM * MODEL_DIM * sizeof(bf16);
  const size_t XBYTES = (size_t)NROWS * MODEL_DIM * sizeof(bf16);
  bf16* Wtq = (bf16*)(ws + off); off += WBYTES;
  bf16* Wtk = (bf16*)(ws + off); off += WBYTES;
  bf16* Wtv = (bf16*)(ws + off); off += WBYTES;
  bf16* Wto = (bf16*)(ws + off); off += WBYTES;
  bf16* qb  = (bf16*)(ws + off); off += XBYTES;
  bf16* kb  = (bf16*)(ws + off); off += XBYTES;
  bf16* vtb = (bf16*)(ws + off); off += XBYTES;   // V transposed: [B][D][L]
  bf16* ctx = (bf16*)(ws + off); off += XBYTES;

  dim3 tgrid(32, 32);
  wconv_kernel<<<tgrid, 256, 0, stream>>>(Wq, Wtq);
  wconv_kernel<<<tgrid, 256, 0, stream>>>(Wk, Wtk);
  wconv_kernel<<<tgrid, 256, 0, stream>>>(Wv, Wtv);
  wconv_kernel<<<tgrid, 256, 0, stream>>>(Wo, Wto);

  dim3 ggrid(MODEL_DIM / 128, NROWS / 128);
  proj_gemm_kernel<0><<<ggrid, 256, 0, stream>>>(query, Wtq, bq, qb);
  proj_gemm_kernel<0><<<ggrid, 256, 0, stream>>>(key_,  Wtk, bk, kb);
  proj_gemm_kernel<1><<<ggrid, 256, 0, stream>>>(value, Wtv, bv, vtb);

  const size_t smem =
      (size_t)(64 * 1032 + 2 * 32 * 1032 + 2 * 256 * 40 + 64 * 40) * sizeof(bf16);  // ~303 KB
  (void)hipFuncSetAttribute((const void*)attn_kernel,
                            hipFuncAttributeMaxDynamicSharedMemorySize, (int)smem);
  attn_kernel<<<dim3(SEQ_L / 64, MODEL_DIM / 256, BATCH), 128, smem, stream>>>(
      qb, kb, vtb, ctx);

  out_gemm_kernel<<<ggrid, 256, 0, stream>>>(ctx, Wto, bo, query, out);
}
